// CrossAttention_18433999635097
// MI455X (gfx1250) — matrix operand order repaired and checked by reading, not compiled, not run on hardware
//
#include <hip/hip_runtime.h>

typedef __attribute__((ext_vector_type(16))) _Float16 v16h;
typedef __attribute__((ext_vector_type(8)))  _Float16 v8h;
typedef __attribute__((ext_vector_type(4)))  _Float16 v4h;
typedef __attribute__((ext_vector_type(8)))  float    v8f;
typedef __attribute__((ext_vector_type(4)))  float    v4f;

#define E_    1024
#define H_    16
#define D_    64
#define LSEQ  2048

static __device__ __forceinline__ v16h cat8(v8h lo, v8h hi) {
  return __builtin_shufflevector(lo, hi, 0,1,2,3,4,5,6,7,8,9,10,11,12,13,14,15);
}
static __device__ __forceinline__ v8f wmma16(v16h a, v16h b, v8f c) {
  // v_wmma_f32_16x16x32_f16 : D = A(16x32) * B(32x16) + C
  return __builtin_amdgcn_wmma_f32_16x16x32_f16(false, a, false, b, (short)0, c, false, false);
}
// async global -> LDS copy, 16B per lane; INST_OFFSET applies to both addresses
static __device__ __forceinline__ void async_b128(unsigned lds_off, const void* gaddr) {
  asm volatile("global_load_async_to_lds_b128 %0, %1, off"
               :: "v"(lds_off), "v"(gaddr) : "memory");
}
static __device__ __forceinline__ void async_b128_o16(unsigned lds_off, const void* gaddr) {
  asm volatile("global_load_async_to_lds_b128 %0, %1, off offset:16"
               :: "v"(lds_off), "v"(gaddr) : "memory");
}
static __device__ __forceinline__ void wait_async0() {
  asm volatile("s_wait_asynccnt 0x0" ::: "memory");
}

// ---------------------------------------------------------------------------
// Elementwise f32 -> f16 (8 elems / thread, b128 stores)
// ---------------------------------------------------------------------------
__global__ __launch_bounds__(256)
void cvt_f16_kernel(const float* __restrict__ src, _Float16* __restrict__ dst, int n8)
{
  const int i = blockIdx.x * 256 + threadIdx.x;
  if (i >= n8) return;
  v4f a = *(const v4f*)(src + (size_t)i * 8);
  v4f b = *(const v4f*)(src + (size_t)i * 8 + 4);
  v8h h;
  h[0] = (_Float16)a[0]; h[1] = (_Float16)a[1]; h[2] = (_Float16)a[2]; h[3] = (_Float16)a[3];
  h[4] = (_Float16)b[0]; h[5] = (_Float16)b[1]; h[6] = (_Float16)b[2]; h[7] = (_Float16)b[3];
  *(v8h*)(dst + (size_t)i * 8) = h;
}

// ---------------------------------------------------------------------------
// W[k][n] f32  ->  Wt[n][k] f16   (64x64 tile via LDS)
// ---------------------------------------------------------------------------
__global__ __launch_bounds__(256)
void transpose_cvt_kernel(const float* __restrict__ W, _Float16* __restrict__ Wt)
{
  __shared__ _Float16 lT[64][72];
  const int t  = threadIdx.x;
  const int r0 = blockIdx.y * 64;
  const int c0 = blockIdx.x * 64;
  const int rr = t >> 2;
  const int cc = (t & 3) * 16;
  #pragma unroll
  for (int j = 0; j < 4; ++j) {
    v4f fv = *(const v4f*)(W + (size_t)(r0 + rr) * E_ + c0 + cc + j * 4);
    lT[rr][cc + j * 4 + 0] = (_Float16)fv[0];
    lT[rr][cc + j * 4 + 1] = (_Float16)fv[1];
    lT[rr][cc + j * 4 + 2] = (_Float16)fv[2];
    lT[rr][cc + j * 4 + 3] = (_Float16)fv[3];
  }
  __syncthreads();
  v8h o0, o1;
  #pragma unroll
  for (int j = 0; j < 8; ++j) { o0[j] = lT[cc + j][rr]; o1[j] = lT[cc + 8 + j][rr]; }
  _Float16* dp = Wt + (size_t)(c0 + rr) * E_ + r0 + cc;
  *(v8h*)dp       = o0;
  *(v8h*)(dp + 8) = o1;
}

// ---------------------------------------------------------------------------
// GEMM: C[4096,1024] = A(f16) * Wt(f16, pre-transposed) + bias
// MODE 0: out16 [B,H,L,D]   MODE 1: out16 [B,H,D,S]   MODE 2: out32 f32
// Double-buffered LDS fed by global_load_async_to_lds_b128 (ASYNCcnt).
// ---------------------------------------------------------------------------
template <int MODE>
__global__ __launch_bounds__(128)
void gemm_proj_kernel(const _Float16* __restrict__ A, const _Float16* __restrict__ Wt,
                      const float* __restrict__ bias,
                      _Float16* __restrict__ out16, float* __restrict__ out32)
{
  __shared__ _Float16 lA[2][64][32];   // [buf][m][k]  4 KB per buf
  __shared__ _Float16 lB[2][64][32];   // [buf][n][k]

  const int t    = threadIdx.x;
  const int lane = t & 31;
  const int wv   = t >> 5;
  const int wm   = wv >> 1;
  const int wn   = wv & 1;
  const int ln   = lane & 15;
  const int g    = lane >> 4;
  const int col0 = blockIdx.x * 64;
  const int row0 = blockIdx.y * 64;

  // staging addresses (fixed per thread): 32 halfs (64B) per thread per tile
  const int sr = t >> 1;              // 0..63
  const int sc = (t & 1) * 16;        // 0 or 16
  const _Float16* ap = A  + (size_t)(row0 + sr) * E_ + sc;
  const _Float16* bp = Wt + (size_t)(col0 + sr) * E_ + sc;
  const unsigned laA = (unsigned)(uintptr_t)&lA[0][sr][sc];
  const unsigned laB = (unsigned)(uintptr_t)&lB[0][sr][sc];
  const unsigned BUFSTRIDE = 64 * 32 * 2;   // bytes per buffer

  v8f acc[2][2];
  #pragma unroll
  for (int i = 0; i < 2; ++i)
    #pragma unroll
    for (int j = 0; j < 2; ++j) acc[i][j] = (v8f)(0.0f);

  // prime k-step 0 into buffer 0
  async_b128    (laA, ap);
  async_b128_o16(laA, ap);
  async_b128    (laB, bp);
  async_b128_o16(laB, bp);

  for (int kt = 0; kt < E_ / 32; ++kt) {
    const int d = kt & 1;
    wait_async0();          // this wave's tile-kt copies have landed in LDS
    __syncthreads();        // everyone's have
    if (kt + 1 < E_ / 32) { // kick off tile kt+1 into the other buffer
      const size_t off = (size_t)(kt + 1) * 32;
      const unsigned nd = (unsigned)((kt + 1) & 1) * BUFSTRIDE;
      async_b128    (laA + nd, ap + off);
      async_b128_o16(laA + nd, ap + off);
      async_b128    (laB + nd, bp + off);
      async_b128_o16(laB + nd, bp + off);
      __builtin_prefetch(ap + off + 64, 0, 0);   // k+2 -> global_prefetch_b8
      __builtin_prefetch(bp + off + 64, 0, 0);
    }
    // fragments + WMMA
    v16h af[2], bf[2];
    #pragma unroll
    for (int mf = 0; mf < 2; ++mf) {
      const int r  = wm * 32 + mf * 16 + ln;
      const int kb = g * 8;
      af[mf] = cat8(*(const v8h*)&lA[d][r][kb], *(const v8h*)&lA[d][r][kb + 16]);
    }
    #pragma unroll
    for (int nf = 0; nf < 2; ++nf) {
      const int c  = wn * 32 + nf * 16 + ln;
      const int kb = g * 8;
      bf[nf] = cat8(*(const v8h*)&lB[d][c][kb], *(const v8h*)&lB[d][c][kb + 16]);
    }
    #pragma unroll
    for (int mf = 0; mf < 2; ++mf)
      #pragma unroll
      for (int nf = 0; nf < 2; ++nf)
        acc[mf][nf] = wmma16(af[mf], bf[nf], acc[mf][nf]);
  }

  // epilogue
  #pragma unroll
  for (int mf = 0; mf < 2; ++mf) {
    #pragma unroll
    for (int nf = 0; nf < 2; ++nf) {
      const int gc   = col0 + wn * 32 + nf * 16 + ln;
      const float bv = bias[gc];
      const int hh   = gc >> 6;
      const int dd   = gc & 63;
      #pragma unroll
      for (int r = 0; r < 8; ++r) {
        const int gr  = row0 + wm * 32 + mf * 16 + g * 8 + r;
        const float v = acc[mf][nf][r] + bv;
        if (MODE == 0) {
          const int bb = gr >> 11, l = gr & 2047;
          out16[(((size_t)bb * H_ + hh) * LSEQ + l) * D_ + dd] = (_Float16)v;
        } else if (MODE == 1) {
          const int bb = gr >> 11, s = gr & 2047;
          out16[(((size_t)bb * H_ + hh) * D_ + dd) * LSEQ + s] = (_Float16)v;
        } else {
          out32[(size_t)gr * E_ + gc] = v;
        }
      }
    }
  }
}

// ---------------------------------------------------------------------------
// Flash attention: Q:[B,H,L,D], K:[B,H,S,D], Vt:[B,H,D,S] (f16) -> O:[B,L,E] f16
// ---------------------------------------------------------------------------
__global__ __launch_bounds__(128)
void attn_kernel(const _Float16* __restrict__ Q, const _Float16* __restrict__ K,
                 const _Float16* __restrict__ Vt, _Float16* __restrict__ O)
{
  __shared__ _Float16 lP[4][16][64];

  const int t    = threadIdx.x;
  const int lane = t & 31;
  const int wv   = t >> 5;
  const int ln   = lane & 15;
  const int g    = lane >> 4;
  const int bh   = blockIdx.y;
  const int b    = bh >> 4;
  const int h    = bh & 15;
  const int q0   = blockIdx.x * 64;

  const _Float16* Qb = Q  + ((size_t)bh * LSEQ + q0) * D_;
  const _Float16* Kb = K  + (size_t)bh * LSEQ * D_;
  const _Float16* Vb = Vt + (size_t)bh * D_ * LSEQ;

  v16h aq[2];
  #pragma unroll
  for (int ks = 0; ks < 2; ++ks) {
    const _Float16* qr = Qb + (size_t)(wv * 16 + ln) * D_ + ks * 32 + g * 8;
    aq[ks] = cat8(*(const v8h*)qr, *(const v8h*)(qr + 16));
  }

  v8f ao[4];
  #pragma unroll
  for (int nf = 0; nf < 4; ++nf) ao[nf] = (v8f)(0.0f);
  float m_i[8], l_i[8];
  #pragma unroll
  for (int r = 0; r < 8; ++r) { m_i[r] = -1e30f; l_i[r] = 0.0f; }

  for (int st = 0; st < LSEQ / 64; ++st) {
    const int s0 = st * 64;

    v8f sc[4];
    #pragma unroll
    for (int nf = 0; nf < 4; ++nf) sc[nf] = (v8f)(0.0f);
    #pragma unroll
    for (int ks = 0; ks < 2; ++ks) {
      #pragma unroll
      for (int nf = 0; nf < 4; ++nf) {
        const int col = s0 + nf * 16 + ln;
        const _Float16* kp = Kb + (size_t)col * D_ + ks * 32 + g * 8;
        v16h bk = cat8(*(const v8h*)kp, *(const v8h*)(kp + 16));
        sc[nf] = wmma16(aq[ks], bk, sc[nf]);
      }
    }
    #pragma unroll
    for (int nf = 0; nf < 4; ++nf)
      #pragma unroll
      for (int r = 0; r < 8; ++r) sc[nf][r] *= 0.125f;

    float mloc[8];
    #pragma unroll
    for (int r = 0; r < 8; ++r) {
      float mx = fmaxf(fmaxf(sc[0][r], sc[1][r]), fmaxf(sc[2][r], sc[3][r]));
      #pragma unroll
      for (int off = 1; off < 16; off <<= 1)
        mx = fmaxf(mx, __shfl_xor(mx, off, 16));
      mloc[r] = mx;
    }
    float sf[8], psum[8];
    #pragma unroll
    for (int r = 0; r < 8; ++r) {
      const float mn = fmaxf(m_i[r], mloc[r]);
      sf[r]   = __expf(m_i[r] - mn);
      m_i[r]  = mn;
      psum[r] = 0.0f;
    }
    #pragma unroll
    for (int nf = 0; nf < 4; ++nf) {
      #pragma unroll
      for (int r = 0; r < 8; ++r) {
        const float p = __expf(sc[nf][r] - m_i[r]);
        lP[wv][g * 8 + r][nf * 16 + ln] = (_Float16)p;
        psum[r] += p;
      }
    }
    #pragma unroll
    for (int r = 0; r < 8; ++r) {
      float s = psum[r];
      #pragma unroll
      for (int off = 1; off < 16; off <<= 1)
        s += __shfl_xor(s, off, 16);
      l_i[r] = l_i[r] * sf[r] + s;
    }
    #pragma unroll
    for (int nf = 0; nf < 4; ++nf)
      #pragma unroll
      for (int r = 0; r < 8; ++r) ao[nf][r] *= sf[r];

    #pragma unroll
    for (int ks = 0; ks < 2; ++ks) {
      const _Float16* pr = &lP[wv][ln][ks * 32 + g * 8];
      v16h ap = cat8(*(const v8h*)pr, *(const v8h*)(pr + 16));
      #pragma unroll
      for (int nf = 0; nf < 4; ++nf) {
        const int d = nf * 16 + ln;
        const _Float16* vp = Vb + (size_t)d * LSEQ + s0 + ks * 32 + g * 8;
        v16h bvf = cat8(*(const v8h*)vp, *(const v8h*)(vp + 16));
        ao[nf] = wmma16(ap, bvf, ao[nf]);
      }
    }
  }

  float inv[8];
  #pragma unroll
  for (int r = 0; r < 8; ++r) inv[r] = 1.0f / l_i[r];
  #pragma unroll
  for (int nf = 0; nf < 4; ++nf) {
    const int d = nf * 16 + ln;
    #pragma unroll
    for (int r = 0; r < 8; ++r) {
      const int row_l = q0 + wv * 16 + g * 8 + r;
      O[((size_t)b * LSEQ + row_l) * E_ + h * D_ + d] = (_Float16)(ao[nf][r] * inv[r]);
    }
  }
}

// ---------------------------------------------------------------------------
extern "C" void kernel_launch(void* const* d_in, const int* in_sizes, int n_in,
                              void* d_out, int out_size, void* d_ws, size_t ws_size,
                              hipStream_t stream) {
  (void)in_sizes; (void)n_in; (void)out_size; (void)ws_size;
  const float* x   = (const float*)d_in[0];
  const float* ctx = (const float*)d_in[1];
  const float* Wq  = (const float*)d_in[2];
  const float* bq  = (const float*)d_in[3];
  const float* Wk  = (const float*)d_in[4];
  const float* bk  = (const float*)d_in[5];
  const float* Wv  = (const float*)d_in[6];
  const float* bv  = (const float*)d_in[7];
  const float* Wp  = (const float*)d_in[8];
  const float* bp  = (const float*)d_in[9];
  float* out = (float*)d_out;

  _Float16* wsh = (_Float16*)d_ws;
  const size_t M4 = (size_t)1 << 22;      // 4M halfs
  const size_t M1 = (size_t)1 << 20;      // 1M halfs
  _Float16* Xh  = wsh;
  _Float16* Ch  = wsh + M4;
  _Float16* Wqt = wsh + 2 * M4;
  _Float16* Wkt = Wqt + M1;
  _Float16* Wvt = Wkt + M1;
  _Float16* Wpt = Wvt + M1;
  _Float16* Qb  = Wpt + M1;
  _Float16* Kb  = Qb + M4;
  _Float16* Vt  = Kb + M4;
  _Float16* Ob  = Vt + M4;

  dim3 thr128(128), thr256(256);
  cvt_f16_kernel<<<dim3((2 * LSEQ * E_ / 8 + 255) / 256), thr256, 0, stream>>>(x,   Xh, 2 * LSEQ * E_ / 8);
  cvt_f16_kernel<<<dim3((2 * LSEQ * E_ / 8 + 255) / 256), thr256, 0, stream>>>(ctx, Ch, 2 * LSEQ * E_ / 8);
  dim3 gtr(E_ / 64, E_ / 64);
  transpose_cvt_kernel<<<gtr, thr256, 0, stream>>>(Wq, Wqt);
  transpose_cvt_kernel<<<gtr, thr256, 0, stream>>>(Wk, Wkt);
  transpose_cvt_kernel<<<gtr, thr256, 0, stream>>>(Wv, Wvt);
  transpose_cvt_kernel<<<gtr, thr256, 0, stream>>>(Wp, Wpt);

  dim3 gblk(E_ / 64, (2 * LSEQ) / 64);
  gemm_proj_kernel<0><<<gblk, thr128, 0, stream>>>(Xh, Wqt, bq, Qb, nullptr);
  gemm_proj_kernel<0><<<gblk, thr128, 0, stream>>>(Ch, Wkt, bk, Kb, nullptr);
  gemm_proj_kernel<1><<<gblk, thr128, 0, stream>>>(Ch, Wvt, bv, Vt, nullptr);
  attn_kernel<<<dim3(LSEQ / 64, 2 * H_), thr128, 0, stream>>>(Qb, Kb, Vt, Ob);
  gemm_proj_kernel<2><<<gblk, thr128, 0, stream>>>(Ob, Wpt, bp, nullptr, out);
}
